// LSTMVanillaStateful_70411693851119
// MI455X (gfx1250) — matrix operand order repaired and checked by reading, not compiled, not run on hardware
//
#include <hip/hip_runtime.h>

typedef _Float16 f16;
typedef _Float16 v8h  __attribute__((ext_vector_type(8)));
typedef _Float16 v16h __attribute__((ext_vector_type(16)));
typedef float    v8f  __attribute__((ext_vector_type(8)));

#define DI 512
#define DH 1024
#define DO 512
#define DB 64
#define DT 512
#define G4H 4096

#if __has_builtin(__builtin_amdgcn_sched_barrier)
#define SCHED_FENCE() __builtin_amdgcn_sched_barrier(0)
#else
#define SCHED_FENCE() ((void)0)
#endif

__device__ __forceinline__ float sigmoid_fast(float x) {
    return 1.0f / (1.0f + __expf(-x));
}
__device__ __forceinline__ float tanh_fast(float x) {
    float e = __expf(-2.0f * x);
    return (1.0f - e) / (1.0f + e);
}

// Build a v16h fragment from two contiguous 16-byte (8 x f16) loads.
__device__ __forceinline__ v16h frag16(const f16* p0, const f16* p1) {
    v8h lo = *(const v8h*)p0;
    v8h hi = *(const v8h*)p1;
    return __builtin_shufflevector(lo, hi, 0,1,2,3,4,5,6,7,8,9,10,11,12,13,14,15);
}

__device__ __forceinline__ v8f wmma_f16(v16h a, v16h b, v8f c) {
    // emits v_wmma_f32_16x16x32_f16
    return __builtin_amdgcn_wmma_f32_16x16x32_f16(
        /*neg_a=*/false, a, /*neg_b=*/false, b,
        /*c_mod=*/(short)0, c, /*reuse_a=*/false, /*reuse_b=*/false);
}

// ---------------------------------------------------------------------------
// Shared GEMM inner loop: one A row stream vs 4 B row streams, KLEN
// contraction, 4 f32 accumulators. Two-stage ping-pong at STAGE granularity
// (1 stage = 2 K-chunks = 20 b128 loads), fenced with sched_barrier(0) so the
// backend scheduler cannot sink the prefetch loads down to their consumers
// (which would serialize every WMMA on a cold L2 access).
// Buffer roles alternate by NAME (no register copies).
// The final trip prefetches one stage past KLEN into the dead buffer (never
// consumed) — callers guarantee that read lands in the workspace (tail pad).
//
// Fragment k order used here (contiguous, same for A and B), arow/bp pre-offset by kb/ko:
//   A (16-bit 16x32): lane holds row m=lane&15, kb=(lane<16)?0:16,
//       elems j = K kb+j (16 contiguous) -> 2x b128.
//   B (16-bit 32x16): lane holds col n=lane&15, ko=(lane<16)?0:16,
//       elems j = K ko+j (16 contiguous) -> 2x b128.
// ---------------------------------------------------------------------------
template <int KLEN>
__device__ __forceinline__ void gemm4_pipe(const f16* __restrict__ arow,
                                           const f16* const bp[4],
                                           v8f acc[4]) {
    v16h a0[2], a1[2], b0[2][4], b1[2][4];

    // Preload stage 0 (chunks 0,1).
#pragma unroll
    for (int q = 0; q < 2; ++q) {
        const int k = q * 32;
        a0[q] = frag16(arow + k, arow + k + 8);
#pragma unroll
        for (int g = 0; g < 4; ++g)
            b0[q][g] = frag16(bp[g] + k, bp[g] + k + 8);
    }
    SCHED_FENCE();

#pragma unroll
    for (int s = 0; s < KLEN / 64; s += 2) {
        // Prefetch stage s+1 into buf1.
#pragma unroll
        for (int q = 0; q < 2; ++q) {
            const int k = (s + 1) * 64 + q * 32;
            a1[q] = frag16(arow + k, arow + k + 8);
#pragma unroll
            for (int g = 0; g < 4; ++g)
                b1[q][g] = frag16(bp[g] + k, bp[g] + k + 8);
        }
        SCHED_FENCE();
        // Compute stage s from buf0.
#pragma unroll
        for (int q = 0; q < 2; ++q)
#pragma unroll
            for (int g = 0; g < 4; ++g)
                acc[g] = wmma_f16(a0[q], b0[q][g], acc[g]);
        SCHED_FENCE();
        // Prefetch stage s+2 into buf0 (over-reads one stage on last trip).
#pragma unroll
        for (int q = 0; q < 2; ++q) {
            const int k = (s + 2) * 64 + q * 32;
            a0[q] = frag16(arow + k, arow + k + 8);
#pragma unroll
            for (int g = 0; g < 4; ++g)
                b0[q][g] = frag16(bp[g] + k, bp[g] + k + 8);
        }
        SCHED_FENCE();
        // Compute stage s+1 from buf1.
#pragma unroll
        for (int q = 0; q < 2; ++q)
#pragma unroll
            for (int g = 0; g < 4; ++g)
                acc[g] = wmma_f16(a1[q], b1[q][g], acc[g]);
        SCHED_FENCE();
    }
}

// ---------------------------------------------------------------------------
// Elementwise conversion helpers
// ---------------------------------------------------------------------------
__global__ void cvt_f32_to_f16(const float* __restrict__ src,
                               f16* __restrict__ dst, int n) {
    int i = blockIdx.x * blockDim.x + threadIdx.x;
    if (i < n) dst[i] = (f16)src[i];
}

__global__ void bias_sum_kernel(const float* __restrict__ b_ih,
                                const float* __restrict__ b_hh,
                                float* __restrict__ bias) {
    int i = blockIdx.x * blockDim.x + threadIdx.x;
    if (i < G4H) bias[i] = b_ih[i] + b_hh[i];
}

// ---------------------------------------------------------------------------
// One LSTM timestep.
// Grid: 64 blocks (one per 16 hidden units), 128 threads (4 waves).
// Wave w handles batch M-tile w (rows 16w..16w+15) and ALL 4 gates for the
// block's 16 hidden units:  gates = x_t @ W_ih^T + h_{t-1} @ W_hh^T + bias,
// so the LSTM cell update is block-local (no cross-block gate exchange).
// Only 256 waves exist in this grid -> occupancy is ~1 wave/SIMD regardless
// of register use; all latency hiding must come from ILP (see gemm4_pipe).
// ---------------------------------------------------------------------------
__global__ __launch_bounds__(128, 1)
void lstm_step_kernel(const f16* __restrict__ seqh,   // [B][T][I] f16
                      const f16* __restrict__ wih,    // [4H][I]   f16
                      const f16* __restrict__ whh,    // [4H][H]   f16
                      const float* __restrict__ bias, // [4H] = b_ih + b_hh
                      float* __restrict__ cbuf,       // [B][H] cell state f32
                      f16* __restrict__ hs,           // [T][B][H] hidden f16
                      int t) {
    const int lane  = threadIdx.x & 31;
    const int mtile = threadIdx.x >> 5;        // 0..3 (batch tile)
    const int u0    = blockIdx.x * 16;         // hidden-unit tile base
    const int n     = lane & 15;
    const int mrow  = mtile * 16 + n;          // A row (batch) this lane loads
    const int kb    = (lane < 16) ? 0 : 16;    // A frag K base
    const int ko    = (lane < 16) ? 0 : 16;    // B frag K base

    v8f acc[4] = {};   // i, f, g, o accumulators (16x16 f32 tiles)

    // ---- Phase 1: input projection, K over I (x_t) ----
    {
        const f16* arow = seqh + ((size_t)mrow * DT + t) * DI + kb;
        const f16* bp[4];
#pragma unroll
        for (int g = 0; g < 4; ++g)
            bp[g] = wih + (size_t)(g * DH + u0 + n) * DI + ko;
        gemm4_pipe<DI>(arow, bp, acc);
    }

    // ---- Phase 2: recurrent projection, K over H (h_{t-1}); h_{-1}=0 ----
    if (t > 0) {
        const f16* arow = hs + ((size_t)(t - 1) * DB + mrow) * DH + kb;
        const f16* bp[4];
#pragma unroll
        for (int g = 0; g < 4; ++g)
            bp[g] = whh + (size_t)(g * DH + u0 + n) * DH + ko;
        gemm4_pipe<DH>(arow, bp, acc);
    }

    // ---- LSTM cell epilogue ----
    // C/D layout: lane holds col n=lane&15; VGPR r holds row r + (lane<16?0:8).
    const int u = u0 + n;
    const float bi = bias[u];
    const float bf = bias[DH + u];
    const float bg = bias[2 * DH + u];
    const float bo = bias[3 * DH + u];
    const int mbase = mtile * 16 + ((lane < 16) ? 0 : 8);
#pragma unroll
    for (int r = 0; r < 8; ++r) {
        const int m = mbase + r;               // batch index
        float ig = sigmoid_fast(acc[0][r] + bi);
        float fg = sigmoid_fast(acc[1][r] + bf);
        float gg = tanh_fast  (acc[2][r] + bg);
        float og = sigmoid_fast(acc[3][r] + bo);
        float cp = (t == 0) ? 0.0f : cbuf[(size_t)m * DH + u];
        float cn = fg * cp + ig * gg;
        cbuf[(size_t)m * DH + u] = cn;
        hs[((size_t)t * DB + m) * DH + u] = (f16)(og * tanh_fast(cn));
    }
}

// ---------------------------------------------------------------------------
// Output projection: out[b][t][o] = hs[t][b][:] . W_out[o][:] + b_out[o]
// Rows r = t*B + b (32768 rows), N = O = 512, K = H = 1024.
// Grid: (256, 8) blocks x 256 threads (8 waves). Wave w owns rows
// r0+16w..r0+16w+15 and a 64-wide column strip (4 N-tiles sharing one A
// fragment -> 4 WMMAs per 10 b128 loads; B reused by all 8 waves via L1).
// ---------------------------------------------------------------------------
__global__ __launch_bounds__(256, 1)
void out_gemm_kernel(const f16* __restrict__ hs,     // [T*B][H] f16
                     const f16* __restrict__ wout,   // [O][H]   f16
                     const float* __restrict__ bout, // [O]
                     float* __restrict__ out) {      // [B][T][O] f32
    const int lane = threadIdx.x & 31;
    const int w    = threadIdx.x >> 5;        // 0..7 = M-tile within block
    const int r0   = blockIdx.x * 128;
    const int n0   = blockIdx.y * 64;
    const int n    = lane & 15;
    const int arow = r0 + w * 16 + n;
    const int kb   = (lane < 16) ? 0 : 16;
    const int ko   = (lane < 16) ? 0 : 16;

    v8f acc[4] = {};
    const f16* aptr = hs + (size_t)arow * DH + kb;
    const f16* bp[4];
#pragma unroll
    for (int j = 0; j < 4; ++j)
        bp[j] = wout + (size_t)(n0 + j * 16 + n) * DH + ko;

    gemm4_pipe<DH>(aptr, bp, acc);

    const int mbase = r0 + w * 16 + ((lane < 16) ? 0 : 8);
#pragma unroll
    for (int j = 0; j < 4; ++j) {
        const int oc = n0 + j * 16 + n;
        const float bv = bout[oc];
#pragma unroll
        for (int r = 0; r < 8; ++r) {
            const int row = mbase + r;         // row = t*B + b
            const int tt = row >> 6;           // / DB
            const int bb = row & 63;           // % DB
            out[((size_t)bb * DT + tt) * DO + oc] = acc[j][r] + bv;
        }
    }
}

// ---------------------------------------------------------------------------
extern "C" void kernel_launch(void* const* d_in, const int* in_sizes, int n_in,
                              void* d_out, int out_size, void* d_ws, size_t ws_size,
                              hipStream_t stream) {
    (void)in_sizes; (void)n_in; (void)out_size; (void)ws_size;

    const float* seq   = (const float*)d_in[0];  // [B,T,I]
    const float* W_ih  = (const float*)d_in[1];  // [4H,I]
    const float* W_hh  = (const float*)d_in[2];  // [4H,H]
    const float* b_ih  = (const float*)d_in[3];  // [4H]
    const float* b_hh  = (const float*)d_in[4];  // [4H]
    const float* W_out = (const float*)d_in[5];  // [O,H]
    const float* b_out = (const float*)d_in[6];  // [O]
    float* out = (float*)d_out;                  // [B,T,O]

    // Carve workspace (assume d_ws is 256B-aligned).
    char* base = (char*)d_ws;
    size_t off = 0;
    auto carve = [&](size_t bytes) -> char* {
        char* r = base + off;
        off = (off + bytes + 255) & ~(size_t)255;
        return r;
    };
    f16*   seqh  = (f16*)  carve((size_t)DB * DT * DI * sizeof(f16));
    f16*   wihh  = (f16*)  carve((size_t)G4H * DI * sizeof(f16));
    f16*   whhh  = (f16*)  carve((size_t)G4H * DH * sizeof(f16));
    f16*   wouth = (f16*)  carve((size_t)DO * DH * sizeof(f16));
    float* bias  = (float*)carve((size_t)G4H * sizeof(float));
    float* cbuf  = (float*)carve((size_t)DB * DH * sizeof(float));
    f16*   hsbuf = (f16*)  carve((size_t)DT * DB * DH * sizeof(f16));
    (void)              carve(256);   // tail pad: pipeline prefetch over-read

    // Precision conversions (once per call; all on stream).
    int n;
    n = DB * DT * DI;
    cvt_f32_to_f16<<<(n + 255) / 256, 256, 0, stream>>>(seq, seqh, n);
    n = G4H * DI;
    cvt_f32_to_f16<<<(n + 255) / 256, 256, 0, stream>>>(W_ih, wihh, n);
    n = G4H * DH;
    cvt_f32_to_f16<<<(n + 255) / 256, 256, 0, stream>>>(W_hh, whhh, n);
    n = DO * DH;
    cvt_f32_to_f16<<<(n + 255) / 256, 256, 0, stream>>>(W_out, wouth, n);
    bias_sum_kernel<<<G4H / 256, 256, 0, stream>>>(b_ih, b_hh, bias);

    // Sequential recurrence: one launch per timestep gives grid-wide sync.
    for (int t = 0; t < DT; ++t) {
        lstm_step_kernel<<<DH / 16, 128, 0, stream>>>(
            seqh, wihh, whhh, bias, cbuf, hsbuf, t);
    }

    // Time-parallel output projection.
    out_gemm_kernel<<<dim3((DB * DT) / 128, DO / 64), 256, 0, stream>>>(
        hsbuf, wouth, b_out, out);
}
